// ScaledDotProductAttention_79886391706128
// MI455X (gfx1250) — matrix operand order repaired and checked by reading, not compiled, not run on hardware
//
#include <hip/hip_runtime.h>

typedef __attribute__((ext_vector_type(16))) _Float16 v16h;
typedef __attribute__((ext_vector_type(4)))  _Float16 v4h;
typedef __attribute__((ext_vector_type(8)))  float    v8f;

#define WMMA16(A, B, C) __builtin_amdgcn_wmma_f32_16x16x32_f16( \
    false, (A), false, (B), (short)0, (C), false, false)

// Problem constants (reference shapes)
constexpr int BATCH = 2;
constexpr int HEADS = 16;
constexpr int SEQ   = 2048;
constexpr int DIM   = 64;

constexpr int QT = 128;   // q rows per workgroup (8 waves x 16)
constexpr int KT = 32;    // k cols per iteration (two 16-wide WMMA tiles)
constexpr int NW = 8;     // waves per workgroup
constexpr float NEG_INF = -1e9f;

__global__ __launch_bounds__(256, 1)
void sdpa_wmma_kernel(const float* __restrict__ Q,
                      const float* __restrict__ K,
                      const float* __restrict__ V,
                      float* __restrict__ ctx,
                      float* __restrict__ attn)
{
    __shared__ _Float16 Qs[QT][72];        // q tile, row-major, padded (144B rows)
    __shared__ _Float16 Ks[KT][72];        // k tile, row-major, padded
    __shared__ _Float16 Vs[DIM][40];       // v tile, d-major (transposed), padded (80B rows)
    __shared__ _Float16 Ps[NW][16][40];    // per-wave prob staging (C->A relayout)

    const int tid  = threadIdx.x;
    const int wave = tid >> 5;
    const int lane = tid & 31;
    const int half = lane >> 4;            // 0: lanes 0-15, 1: lanes 16-31
    const int l16  = lane & 15;

    const int bh    = blockIdx.x / (SEQ / QT);
    const int qblk  = blockIdx.x % (SEQ / QT);
    const int qbase = qblk * QT;

    const float* Qb = Q + (size_t)bh * SEQ * DIM;
    const float* Kb = K + (size_t)bh * SEQ * DIM;
    const float* Vb = V + (size_t)bh * SEQ * DIM;

    // ---- stage Q tile to LDS as f16 (float4 loads, packed f16x4 LDS stores) ----
    for (int i = tid; i < QT * DIM / 4; i += 256) {
        int r = i >> 4, c = (i & 15) << 2;
        float4 q4 = reinterpret_cast<const float4*>(Qb)[(size_t)(qbase + r) * 16 + (i & 15)];
        v4h h = { (_Float16)q4.x, (_Float16)q4.y, (_Float16)q4.z, (_Float16)q4.w };
        *reinterpret_cast<v4h*>(&Qs[r][c]) = h;
    }
    __syncthreads();

    // ---- build A operands (16x32 f16 layout; M = lane%16) ----
    const int aq = wave * 16 + l16;
    v16h a_lo, a_hi;
#pragma unroll
    for (int e = 0; e < 16; ++e) {
        int d = ((e >> 3) << 4) + (half << 3) + (e & 7);   // contraction index in A layout
        a_lo[e] = Qs[aq][d];
        a_hi[e] = Qs[aq][d + 32];
    }

    // Per-lane online softmax stats over this lane's own columns only.
    // Lane contributes to rows (r + 8*half), columns {kb+l16, kb+16+l16}.
    float m[8], l[8];
#pragma unroll
    for (int r = 0; r < 8; ++r) { m[r] = -3.0e38f; l[r] = 0.0f; }

    // ======== PASS 1: per-lane online (m,l); no cross-lane ops in loop ========
    for (int kb = 0; kb < SEQ; kb += KT) {
        __syncthreads();
        for (int i = tid; i < KT * DIM / 4; i += 256) {
            int r = i >> 4, c = (i & 15) << 2;
            float4 k4 = reinterpret_cast<const float4*>(Kb)[(size_t)(kb + r) * 16 + (i & 15)];
            v4h h = { (_Float16)k4.x, (_Float16)k4.y, (_Float16)k4.z, (_Float16)k4.w };
            *reinterpret_cast<v4h*>(&Ks[r][c]) = h;
        }
        if (kb + KT < SEQ) {   // prefetch next K tile (global_prefetch_b8)
            const char* nk = (const char*)(Kb + (size_t)(kb + KT) * DIM);
            __builtin_prefetch(nk + tid * 32, 0, 1);
        }
        __syncthreads();

        // B operands: n = l16, element e -> contraction index d = 16*(e>>3) + 8*half + (e&7)
        v16h b0l, b0h, b1l, b1h;
#pragma unroll
        for (int e = 0; e < 16; ++e) {
            int d = ((e >> 3) << 4) + (half << 3) + (e & 7);
            b0l[e] = Ks[l16][d];        b0h[e] = Ks[l16][d + 32];
            b1l[e] = Ks[16 + l16][d];   b1h[e] = Ks[16 + l16][d + 32];
        }

        v8f c0 = {}, c1 = {};
        c0 = WMMA16(a_lo, b0l, c0);  c0 = WMMA16(a_hi, b0h, c0);
        c1 = WMMA16(a_lo, b1l, c1);  c1 = WMMA16(a_hi, b1h, c1);

#pragma unroll
        for (int r = 0; r < 8; ++r) {
            float s0 = c0[r]; s0 = (s0 == 0.0f) ? NEG_INF : s0;
            float s1 = c1[r]; s1 = (s1 == 0.0f) ? NEG_INF : s1;
            float mn = fmaxf(m[r], fmaxf(s0, s1));
            l[r] = l[r] * __expf(m[r] - mn) + __expf(s0 - mn) + __expf(s1 - mn);
            m[r] = mn;
        }
    }

    // ---- single cross-lane (m,l) merge: 16 lanes within each half ----
    float inv_l[8];
#pragma unroll
    for (int r = 0; r < 8; ++r) {
#pragma unroll
        for (int off = 1; off <= 8; off <<= 1) {
            float mo = __shfl_xor(m[r], off, 32);
            float lo = __shfl_xor(l[r], off, 32);
            float mn = fmaxf(m[r], mo);
            l[r] = l[r] * __expf(m[r] - mn) + lo * __expf(mo - mn);
            m[r] = mn;
        }
        inv_l[r] = 1.0f / l[r];
    }

    // ======== PASS 2: recompute, emit attn (non-temporal), O = P @ V ========
    v8f o[4] = {{}, {}, {}, {}};

    for (int kb = 0; kb < SEQ; kb += KT) {
        __syncthreads();
        for (int i = tid; i < KT * DIM / 4; i += 256) {
            int r = i >> 4, c = (i & 15) << 2;
            float4 k4 = reinterpret_cast<const float4*>(Kb)[(size_t)(kb + r) * 16 + (i & 15)];
            v4h h = { (_Float16)k4.x, (_Float16)k4.y, (_Float16)k4.z, (_Float16)k4.w };
            *reinterpret_cast<v4h*>(&Ks[r][c]) = h;
            float4 v4 = reinterpret_cast<const float4*>(Vb)[(size_t)(kb + r) * 16 + (i & 15)];
            Vs[c + 0][r] = (_Float16)v4.x;   // d-major for contiguous B-operand reads
            Vs[c + 1][r] = (_Float16)v4.y;
            Vs[c + 2][r] = (_Float16)v4.z;
            Vs[c + 3][r] = (_Float16)v4.w;
        }
        if (kb + KT < SEQ) {   // prefetch next K and V tiles
            const char* nk = (const char*)(Kb + (size_t)(kb + KT) * DIM);
            const char* nv = (const char*)(Vb + (size_t)(kb + KT) * DIM);
            __builtin_prefetch(nk + tid * 32, 0, 1);
            __builtin_prefetch(nv + tid * 32, 0, 1);
        }
        __syncthreads();

        v16h b0l, b0h, b1l, b1h;
#pragma unroll
        for (int e = 0; e < 16; ++e) {
            int d = ((e >> 3) << 4) + (half << 3) + (e & 7);
            b0l[e] = Ks[l16][d];        b0h[e] = Ks[l16][d + 32];
            b1l[e] = Ks[16 + l16][d];   b1h[e] = Ks[16 + l16][d + 32];
        }

        v8f c0 = {}, c1 = {};
        c0 = WMMA16(a_lo, b0l, c0);  c0 = WMMA16(a_hi, b0h, c0);
        c1 = WMMA16(a_lo, b1l, c1);  c1 = WMMA16(a_hi, b1h, c1);

#pragma unroll
        for (int r = 0; r < 8; ++r) {
            float s0 = c0[r]; s0 = (s0 == 0.0f) ? NEG_INF : s0;
            float s1 = c1[r]; s1 = (s1 == 0.0f) ? NEG_INF : s1;
            float p0 = __expf(s0 - m[r]) * inv_l[r];
            float p1 = __expf(s1 - m[r]) * inv_l[r];
            int row = r + 8 * half;                       // C/D layout row
            size_t arow = ((size_t)bh * SEQ + (qbase + wave * 16 + row)) * SEQ + kb;
            __builtin_nontemporal_store(p0, &attn[arow + l16]);        // 537MB stream: NT
            __builtin_nontemporal_store(p1, &attn[arow + 16 + l16]);
            Ps[wave][row][l16]      = (_Float16)p0;
            Ps[wave][row][16 + l16] = (_Float16)p1;
        }
        __syncthreads();   // order Ps writes before relayout reads (uniform)

        // Relayout P (C layout -> A layout): M = l16, element e -> k index
        v16h ap;
#pragma unroll
        for (int e = 0; e < 16; ++e)
            ap[e] = Ps[wave][l16][((e >> 3) << 4) + (half << 3) + (e & 7)];

        // O[g] += P(16x32) x V(32x16), g selects 16-wide d slice
#pragma unroll
        for (int g = 0; g < 4; ++g) {
            v16h bv;
#pragma unroll
            for (int e = 0; e < 16; ++e)
                bv[e] = Vs[g * 16 + l16][((e >> 3) << 4) + (half << 3) + (e & 7)];
            o[g] = WMMA16(ap, bv, o[g]);
        }
    }

    // ---- write context ----
#pragma unroll
    for (int g = 0; g < 4; ++g) {
#pragma unroll
        for (int r = 0; r < 8; ++r) {
            int row = r + 8 * half;
            ctx[((size_t)bh * SEQ + (qbase + wave * 16 + row)) * DIM + g * 16 + l16] = o[g][r];
        }
    }
}

extern "C" void kernel_launch(void* const* d_in, const int* in_sizes, int n_in,
                              void* d_out, int out_size, void* d_ws, size_t ws_size,
                              hipStream_t stream) {
    (void)in_sizes; (void)n_in; (void)d_ws; (void)ws_size; (void)out_size;
    const float* Q = (const float*)d_in[0];
    const float* K = (const float*)d_in[1];
    const float* V = (const float*)d_in[2];
    float* ctx  = (float*)d_out;
    float* attn = ctx + (size_t)BATCH * HEADS * SEQ * DIM;   // tuple order: (context, attn)

    dim3 grid(BATCH * HEADS * (SEQ / QT));
    dim3 block(256);
    sdpa_wmma_kernel<<<grid, block, 0, stream>>>(Q, K, V, ctx, attn);
}
